// HierarchicalConversationGNN_5549097747001
// MI455X (gfx1250) — matrix operand order repaired and checked by reading, not compiled, not run on hardware
//
#include <hip/hip_runtime.h>

// ============================================================================
// HierarchicalConversationGNN forward — MI455X / gfx1250 (wave32, WMMA).
//
// Strategy:
//  * All dense GEMMs (~24 GFLOP, the dominant cost; working set is L2-resident
//    at 192 MB so this is compute-bound) run through v_wmma_f32_16x16x32_bf16:
//    bf16 operands, f32 accumulation. Each wave owns a 16x64 output strip
//    (1 A-fragment reused by 4 B-fragments -> 4 WMMAs per A load). The K-loop
//    is a compile-time constant (128 or 256) and fully unrolled so the
//    scheduler hoists fragment loads ahead of WMMAs (no loadcnt-0 stall per
//    WMMA) and folds addresses into instruction offsets.
//  * Token/message "graphs" are fixed stencils (j +/- w within a sequence;
//    chain for messages), so GAT softmax-aggregation is a per-node wave
//    kernel: no atomics, no segment ops, no edge lists.
//  * Edge features have only 7 (token) / 65 (message) distinct rows ->
//    precomputed tiny tables (e_vec, s_edge).
//  * Cross-attention top-12: per-head K cached in LDS (odd dword row stride
//    -> bank-conflict-free), q broadcast via __shfl, exact iterative top-k.
//
// Input flattening assumption: setup_inputs() dict insertion order,
// nested dicts depth-first in insertion order (58 leaves, enum below).
// Workspace assumption: ws_size >= ~70 MB.
// ============================================================================

typedef unsigned short u16;
typedef unsigned int   u32;
typedef __attribute__((ext_vector_type(16))) __bf16 v16bf;
typedef __attribute__((ext_vector_type(8)))  float  v8f;
typedef __attribute__((ext_vector_type(4)))  u32    u32x4;

#define HEADS 4
#define MSGS  128
#define SEQ   256
#define NTOK  (MSGS*SEQ)   // 32768
#define HID   256
#define DTOKD 128
#define DMSGD 128

// ---- input index map (insertion-order flattening of setup_inputs()) --------
enum {
  IN_TOK_EMB = 0, IN_POS_EMB, IN_SEG_EMB, IN_LN_EMB_S, IN_LN_EMB_B, IN_TOK_EDGE_EMB,
  IN_T0_W, IN_T0_B, IN_T0_WE, IN_T0_ASRC, IN_T0_ADST, IN_T0_AEDGE, IN_T0_LNS, IN_T0_LNB,
  IN_T1_W, IN_T1_B, IN_T1_WE, IN_T1_ASRC, IN_T1_ADST, IN_T1_AEDGE, IN_T1_LNS, IN_T1_LNB,
  IN_CR_WQ, IN_CR_WK, IN_CR_WV, IN_CR_WO,
  IN_WK2, IN_WV2, IN_Q2, IN_POSW, IN_POSB, IN_WO2, IN_BO2,
  IN_ENC_W, IN_ENC_B, IN_ENC_LNS, IN_ENC_LNB,
  IN_MSG_RELPOS,
  IN_M0_W, IN_M0_B, IN_M0_WE, IN_M0_ASRC, IN_M0_ADST, IN_M0_AEDGE, IN_M0_LNS, IN_M0_LNB,
  IN_M1_W, IN_M1_B, IN_M1_WE, IN_M1_ASRC, IN_M1_ADST, IN_M1_AEDGE, IN_M1_LNS, IN_M1_LNB,
  IN_TOKEN_IDS, IN_SEGMENT_IDS, IN_MSG_ATTR, IN_MSG_EDGE_INDEX
};

// ---- bf16 bit helpers (avoid scalar __bf16 arithmetic; RNE rounding) -------
__device__ __forceinline__ float bf2f(u16 b) {
  union { u32 u; float f; } x; x.u = ((u32)b) << 16; return x.f;
}
__device__ __forceinline__ u16 f2bf(float f) {
  union { float f; u32 u; } x; x.f = f;
  return (u16)((x.u + 0x7fffu + ((x.u >> 16) & 1u)) >> 16);
}
__device__ __forceinline__ float wsum(float v) {
  #pragma unroll
  for (int o = 16; o; o >>= 1) v += __shfl_xor(v, o, 32);
  return v;
}
__device__ __forceinline__ float wmaxr(float v) {
  #pragma unroll
  for (int o = 16; o; o >>= 1) v = fmaxf(v, __shfl_xor(v, o, 32));
  return v;
}

union Frag { v16bf v; u32x4 q[2]; };

// ============================================================================
// bf16 WMMA GEMM: Out(R,C) = A(R,KK) @ B(KK,C) [+bias] [+resid], bf16 out.
// BT is B transposed (C,KK) row-major so both fragments load contiguously.
// One wave = 16x64 output strip (4 accumulators share one A fragment).
// KK is a template constant (128/256): full unroll -> loads scheduled ahead
// of WMMAs, fragment addresses folded into 24-bit instruction offsets.
// ============================================================================
template <int KK>
__global__ void gemm_bf16_wmma(const u16* __restrict__ A, const u16* __restrict__ BT,
                               const float* __restrict__ bias, const u16* __restrict__ resid,
                               u16* __restrict__ Out, int R, int C)
{
  const int lane   = threadIdx.x & 31;
  const int wid    = blockIdx.x * (blockDim.x >> 5) + (threadIdx.x >> 5);
  const int tilesC = C >> 6;                 // 64-column groups
  const int tileM  = wid / tilesC;
  const int tileN  = wid - tileM * tilesC;
  if (tileM * 16 >= R) return;
  const int hi = lane >> 4, lr = lane & 15;

  const u16* aRow = A  + (size_t)(tileM * 16 + lr) * KK;
  const u16* bRow = BT + (size_t)(tileN * 64 + lr) * KK;
  __builtin_prefetch(aRow + KK, 0, 1);       // global_prefetch_b8: next row panel

  v8f acc[4] = {v8f{}, v8f{}, v8f{}, v8f{}};
  #pragma unroll
  for (int k0 = 0; k0 < KK; k0 += 32) {
    Frag a;
    // A 16x32 bf16 fragment (ISA 7.12.2): lanes 0-15: K {k0..k0+7, k0+16..k0+23}
    a.q[0] = *(const u32x4*)(aRow + k0 + hi * 8);
    a.q[1] = *(const u32x4*)(aRow + k0 + 16 + hi * 8);
    #pragma unroll
    for (int t = 0; t < 4; ++t) {
      Frag b;
      const u16* bp = bRow + t * 16 * KK + k0 + hi * 8;
      b.q[0] = *(const u32x4*)(bp);
      b.q[1] = *(const u32x4*)(bp + 16);
      acc[t] = __builtin_amdgcn_wmma_f32_16x16x32_bf16(
          false, a.v, false, b.v, (short)0, acc[t], false, false);
    }
  }
  const bool hasB = (bias != nullptr), hasR = (resid != nullptr);
  #pragma unroll
  for (int t = 0; t < 4; ++t) {
    int n = tileN * 64 + t * 16 + (lane & 15);
    float bv = hasB ? bias[n] : 0.f;
    #pragma unroll
    for (int r = 0; r < 8; ++r) {
      int m = tileM * 16 + r + 8 * hi;       // C/D layout: m = r + 8*hi, n = lane&15
      float val = acc[t][r] + bv;
      if (hasR) val += bf2f(resid[(size_t)m * C + n]);
      Out[(size_t)m * C + n] = f2bf(val);
    }
  }
}

// ---- weight transpose + f32->bf16 conversion (run once per weight) ---------
__global__ void convert_wT(const float* __restrict__ W, u16* __restrict__ BT, int K, int C)
{
  int idx = blockIdx.x * blockDim.x + threadIdx.x;
  if (idx >= K * C) return;
  int k = idx / C, c = idx - k * C;
  BT[(size_t)c * K + k] = f2bf(W[idx]);
}

// ---- embedding + LN -> bf16 (one wave per token) ---------------------------
__global__ void embed_kernel(const float* __restrict__ tok_emb, const float* __restrict__ pos_emb,
                             const float* __restrict__ seg_emb, const float* __restrict__ ln_s,
                             const float* __restrict__ ln_b, const int* __restrict__ token_ids,
                             const int* __restrict__ segment_ids, u16* __restrict__ out)
{
  int node = blockIdx.x * (blockDim.x >> 5) + (threadIdx.x >> 5);
  if (node >= NTOK) return;
  int lane = threadIdx.x & 31;
  int l = node & (SEQ - 1);
  int tid = token_ids[node], sid = segment_ids[node];
  float v[4];
  #pragma unroll
  for (int i = 0; i < 4; ++i) {
    int d = i * 32 + lane;
    v[i] = tok_emb[(size_t)tid * DTOKD + d] + pos_emb[l * DTOKD + d] + seg_emb[sid * DTOKD + d];
  }
  float s = 0.f, s2 = 0.f;
  #pragma unroll
  for (int i = 0; i < 4; ++i) { s += v[i]; s2 += v[i] * v[i]; }
  s = wsum(s); s2 = wsum(s2);
  float mean = s * (1.f / DTOKD), var = s2 * (1.f / DTOKD) - mean * mean;
  float inv = rsqrtf(var + 1e-5f);
  #pragma unroll
  for (int i = 0; i < 4; ++i) {
    int d = i * 32 + lane;
    out[(size_t)node * DTOKD + d] = f2bf((v[i] - mean) * inv * ln_s[d] + ln_b[d]);
  }
}

// ---- per-node attention scalars s_src = h.a_src, s_dst = h.a_dst -----------
__global__ void attn_score_kernel(const u16* __restrict__ h, const float* __restrict__ a_src,
                                  const float* __restrict__ a_dst, float* __restrict__ s_src,
                                  float* __restrict__ s_dst, int n_nodes, int O)
{
  int node = blockIdx.x * (blockDim.x >> 5) + (threadIdx.x >> 5);
  if (node >= n_nodes) return;
  int lane = threadIdx.x & 31;
  int dh = O >> 2;
  const u16* row = h + (size_t)node * O;
  for (int hd = 0; hd < HEADS; ++hd) {
    float ps = 0.f, pd = 0.f;
    for (int d = lane; d < dh; d += 32) {
      float x = bf2f(row[hd * dh + d]);
      ps += x * a_src[hd * dh + d];
      pd += x * a_dst[hd * dh + d];
    }
    ps = wsum(ps); pd = wsum(pd);
    if (lane == 0) { s_src[node * 4 + hd] = ps; s_dst[node * 4 + hd] = pd; }
  }
}

// ---- distinct edge features: e[r] = emb[r]@We, s_edge[r,h] = e.a_edge ------
__global__ void edge_embed_kernel(const float* __restrict__ emb, const float* __restrict__ We,
                                  const float* __restrict__ a_edge, float* __restrict__ e_out,
                                  float* __restrict__ s_edge, int O)
{
  __shared__ float es[256];
  int r = blockIdx.x, t = threadIdx.x;   // blockDim == O
  float acc = 0.f;
  for (int i = 0; i < 64; ++i) acc += emb[r * 64 + i] * We[i * O + t];
  es[t] = acc;
  e_out[r * O + t] = acc;
  __syncthreads();
  int dh = O >> 2;
  if (t < HEADS) {
    float s = 0.f;
    for (int d = 0; d < dh; ++d) s += es[t * dh + d] * a_edge[t * dh + d];
    s_edge[r * 4 + t] = s;
  }
}

// ---- GAT stencil: softmax over <=2*MAXW neighbors, aggregate, LN, ELU ------
template <int O, int MAXW>
__global__ void gat_stencil_kernel(const u16* __restrict__ h, const float* __restrict__ s_src,
                                   const float* __restrict__ s_dst, const float* __restrict__ e_vec,
                                   const float* __restrict__ s_edge, const u16* __restrict__ resid,
                                   const float* __restrict__ ln_s, const float* __restrict__ ln_b,
                                   u16* __restrict__ out_bf, float* __restrict__ out_f32,
                                   int n_nodes, int seqLen, int relBase)
{
  constexpr int DPL = O / 32, DH = O / 4, NB = 2 * MAXW;
  int node = blockIdx.x * (blockDim.x >> 5) + (threadIdx.x >> 5);
  if (node >= n_nodes) return;
  int lane = threadIdx.x & 31;
  int l = node & (seqLen - 1);             // seqLen is a power of two (256 / 128)

  int src[NB], rel[NB]; bool ok[NB];
  #pragma unroll
  for (int w = 1; w <= MAXW; ++w) {
    int a = 2 * (w - 1);
    ok[a]     = (l >= w);           src[a]     = ok[a]     ? node - w : node; rel[a]     = relBase + w;
    ok[a + 1] = (l + w < seqLen);   src[a + 1] = ok[a + 1] ? node + w : node; rel[a + 1] = relBase - w;
  }

  float alpha[NB][HEADS];
  #pragma unroll
  for (int hd = 0; hd < HEADS; ++hd) {
    float sdn = s_dst[node * 4 + hd];
    float lg[NB]; float mx = -3.0e38f;
    #pragma unroll
    for (int j = 0; j < NB; ++j) {
      float x = -3.0e38f;
      if (ok[j]) {
        x = s_src[src[j] * 4 + hd] + sdn + s_edge[rel[j] * 4 + hd];
        x = x > 0.f ? x : 0.2f * x;        // leaky_relu(0.2)
      }
      lg[j] = x; mx = fmaxf(mx, x);
    }
    float den = 0.f;
    #pragma unroll
    for (int j = 0; j < NB; ++j) {
      float e = ok[j] ? expf(lg[j] - mx) : 0.f;
      lg[j] = e; den += e;
    }
    float inv = 1.f / (den + 1e-16f);
    #pragma unroll
    for (int j = 0; j < NB; ++j) alpha[j][hd] = lg[j] * inv;
  }

  float acc[DPL];
  #pragma unroll
  for (int i = 0; i < DPL; ++i) acc[i] = 0.f;
  #pragma unroll
  for (int j = 0; j < NB; ++j) {
    const u16* hr = h + (size_t)src[j] * O;
    const float* er = e_vec + rel[j] * O;
    #pragma unroll
    for (int i = 0; i < DPL; ++i) {
      int d = i * 32 + lane;
      acc[i] += alpha[j][(i * 32) / DH] * (bf2f(hr[d]) + er[d]);   // head idx lane-invariant
    }
  }
  if (resid) {
    const u16* rr = resid + (size_t)node * O;
    #pragma unroll
    for (int i = 0; i < DPL; ++i) acc[i] += bf2f(rr[i * 32 + lane]);
  }
  float s = 0.f, s2 = 0.f;
  #pragma unroll
  for (int i = 0; i < DPL; ++i) { s += acc[i]; s2 += acc[i] * acc[i]; }
  s = wsum(s); s2 = wsum(s2);
  float mean = s * (1.f / O), var = s2 * (1.f / O) - mean * mean;
  float inv = rsqrtf(var + 1e-5f);
  #pragma unroll
  for (int i = 0; i < DPL; ++i) {
    int d = i * 32 + lane;
    float v = (acc[i] - mean) * inv * ln_s[d] + ln_b[d];
    v = v > 0.f ? v : expm1f(v);                       // ELU
    if (out_bf)  out_bf[(size_t)node * O + d] = f2bf(v);
    if (out_f32) out_f32[(size_t)node * O + d] = v;
  }
}

// ---- per-sequence mean over L tokens -> msg (M,256) ------------------------
__global__ void seq_mean_kernel(const u16* __restrict__ x, u16* __restrict__ msg)
{
  int m = blockIdx.x, d = threadIdx.x;   // blockDim = 256
  const u16* base = x + (size_t)m * SEQ * HID + d;
  float s = 0.f;
  for (int l = 0; l < SEQ; ++l) s += bf2f(base[(size_t)l * HID]);
  msg[m * HID + d] = f2bf(s * (1.f / SEQ));
}

// ---- cross attention: per-head K in LDS, exact top-12, softmax, gather -----
__global__ void cross_topk_kernel(const u16* __restrict__ q, const u16* __restrict__ k,
                                  const u16* __restrict__ v, u16* __restrict__ outattn, int Nn)
{
  __shared__ u32 ksd[128 * 33];          // 64 bf16/row, 33-dword stride (odd -> no bank conflicts)
  const int lane = threadIdx.x & 31, wave = threadIdx.x >> 5, waves = blockDim.x >> 5;
  const int gw = blockIdx.x * waves + wave, gws = gridDim.x * waves;
  const u32* qU = (const u32*)q; const u32* kU = (const u32*)k; const u32* vU = (const u32*)v;
  u32* oU = (u32*)outattn;

  for (int hd = 0; hd < HEADS; ++hd) {
    __syncthreads();
    for (int idx = threadIdx.x; idx < 128 * 32; idx += blockDim.x) {
      int row = idx >> 5, c = idx & 31;
      ksd[row * 33 + c] = kU[row * 128 + hd * 32 + c];
    }
    __syncthreads();
    for (int n = gw; n < Nn; n += gws) {
      u32 qp = qU[(size_t)n * 128 + hd * 32 + lane];
      int mown = n / SEQ;
      float sc[4];
      #pragma unroll
      for (int c = 0; c < 4; ++c) {
        int m = lane + 32 * c;
        const u32* kr = &ksd[m * 33];
        float s = 0.f;
        #pragma unroll 8
        for (int d = 0; d < 32; ++d) {
          u32 qq = __shfl(qp, d, 32);
          u32 kk = kr[d];
          s += bf2f((u16)(qq & 0xffffu)) * bf2f((u16)(kk & 0xffffu))
             + bf2f((u16)(qq >> 16))     * bf2f((u16)(kk >> 16));
        }
        s *= 0.125f;                     // 1/sqrt(64)
        if (m == mown) s = -1.0e9f;      // mask own message
        sc[c] = s;
      }
      float wv[12]; int wi[12];
      #pragma unroll
      for (int t = 0; t < 12; ++t) {     // K = int(0.1*128) = 12
        float best = sc[0]; int bi = lane;
        #pragma unroll
        for (int c = 1; c < 4; ++c) { int m = lane + 32 * c; if (sc[c] > best) { best = sc[c]; bi = m; } }
        #pragma unroll
        for (int o = 16; o; o >>= 1) {
          float ov = __shfl_xor(best, o, 32); int oi = __shfl_xor(bi, o, 32);
          if (ov > best || (ov == best && oi < bi)) { best = ov; bi = oi; }
        }
        wv[t] = best; wi[t] = bi;
        if ((bi & 31) == lane) {         // invalidate chosen slot (no dynamic reg index)
          int slot = bi >> 5;
          if (slot == 0) sc[0] = -3.0e38f; else if (slot == 1) sc[1] = -3.0e38f;
          else if (slot == 2) sc[2] = -3.0e38f; else sc[3] = -3.0e38f;
        }
      }
      float mx = wv[0], den = 0.f;
      #pragma unroll
      for (int t = 0; t < 12; ++t) { wv[t] = expf(wv[t] - mx); den += wv[t]; }
      float inv = 1.f / den, o0 = 0.f, o1 = 0.f;
      #pragma unroll
      for (int t = 0; t < 12; ++t) {
        float w = wv[t] * inv;
        u32 vv = vU[(size_t)wi[t] * 128 + hd * 32 + lane];
        o0 += w * bf2f((u16)(vv & 0xffffu));
        o1 += w * bf2f((u16)(vv >> 16));
      }
      oU[(size_t)n * 128 + hd * 32 + lane] = (u32)f2bf(o0) | ((u32)f2bf(o1) << 16);
    }
  }
}

// ---- token_to_message: per-(m,head) softmax over L, attention pooling ------
__global__ void t2m_kernel(const u16* __restrict__ k2, const u16* __restrict__ v2,
                           const float* __restrict__ q2, const float* __restrict__ pos_w,
                           const float* __restrict__ pos_b, u16* __restrict__ pooled)
{
  __shared__ float a_sh[HEADS][SEQ];
  int m = blockIdx.x;
  int lane = threadIdx.x & 31, hd = threadIdx.x >> 5;   // blockDim = 128 -> wave per head
  const u32* kU = (const u32*)k2; const u32* vU = (const u32*)v2;
  float pw = pos_w[hd], pb = pos_b[hd];
  float s[8];
  #pragma unroll
  for (int c = 0; c < 8; ++c) {
    int l = c * 32 + lane;
    const u32* kr = kU + ((size_t)(m * SEQ + l)) * 128 + hd * 32;
    float acc = 0.f;
    for (int d = 0; d < 32; ++d) {
      u32 kk = kr[d];
      acc += bf2f((u16)(kk & 0xffffu)) * q2[hd * 64 + 2 * d]
           + bf2f((u16)(kk >> 16))     * q2[hd * 64 + 2 * d + 1];
    }
    s[c] = acc * 0.125f + ((float)l * (1.f / SEQ)) * pw + pb;
  }
  float mx = -3.0e38f;
  #pragma unroll
  for (int c = 0; c < 8; ++c) mx = fmaxf(mx, s[c]);
  mx = wmaxr(mx);
  float den = 0.f;
  #pragma unroll
  for (int c = 0; c < 8; ++c) { s[c] = expf(s[c] - mx); den += s[c]; }
  den = wsum(den);
  float inv = 1.f / den;
  #pragma unroll
  for (int c = 0; c < 8; ++c) a_sh[hd][c * 32 + lane] = s[c] * inv;
  __syncthreads();
  float o0 = 0.f, o1 = 0.f;
  for (int l = 0; l < SEQ; ++l) {
    float a = a_sh[hd][l];
    u32 vv = vU[((size_t)(m * SEQ + l)) * 128 + hd * 32 + lane];
    o0 += a * bf2f((u16)(vv & 0xffffu));
    o1 += a * bf2f((u16)(vv >> 16));
  }
  ((u32*)pooled)[m * 128 + hd * 32 + lane] = (u32)f2bf(o0) | ((u32)f2bf(o1) << 16);
}

// ---- encoder: [msg, attr] @ enc_W + b, LN, ReLU (K=130 handled here) -------
__global__ void enc_kernel(const u16* __restrict__ msgout, const float* __restrict__ attr,
                           const float* __restrict__ W, const float* __restrict__ b,
                           const float* __restrict__ lns, const float* __restrict__ lnb,
                           u16* __restrict__ feat)
{
  int m = blockIdx.x * (blockDim.x >> 5) + (threadIdx.x >> 5);
  if (m >= MSGS) return;
  int lane = threadIdx.x & 31;
  float acc[4] = {0.f, 0.f, 0.f, 0.f};
  for (int i = 0; i < DMSGD; ++i) {
    float x = bf2f(msgout[m * DMSGD + i]);
    #pragma unroll
    for (int j = 0; j < 4; ++j) acc[j] += x * W[i * DMSGD + j * 32 + lane];
  }
  float a0 = attr[m * 2], a1 = attr[m * 2 + 1];
  #pragma unroll
  for (int j = 0; j < 4; ++j) {
    int d = j * 32 + lane;
    acc[j] += a0 * W[128 * DMSGD + d] + a1 * W[129 * DMSGD + d] + b[d];
  }
  float s = 0.f, s2 = 0.f;
  #pragma unroll
  for (int j = 0; j < 4; ++j) { s += acc[j]; s2 += acc[j] * acc[j]; }
  s = wsum(s); s2 = wsum(s2);
  float mean = s * (1.f / DMSGD), var = s2 * (1.f / DMSGD) - mean * mean;
  float inv = rsqrtf(var + 1e-5f);
  #pragma unroll
  for (int j = 0; j < 4; ++j) {
    int d = j * 32 + lane;
    float v = (acc[j] - mean) * inv * lns[d] + lnb[d];
    feat[m * DMSGD + d] = f2bf(fmaxf(v, 0.f));
  }
}

// ============================================================================
extern "C" void kernel_launch(void* const* d_in, const int* in_sizes, int n_in,
                              void* d_out, int out_size, void* d_ws, size_t ws_size,
                              hipStream_t stream)
{
  (void)in_sizes; (void)n_in; (void)out_size; (void)ws_size;
  auto F = [&](int i) { return (const float*)d_in[i]; };
  auto I = [&](int i) { return (const int*)d_in[i]; };

  // ---- workspace (bump allocator, 256B aligned; ~70 MB total) -------------
  char* base = (char*)d_ws; size_t off = 0;
  auto alloc = [&](size_t b) -> void* {
    void* p = base + off; off = (off + b + 255) & ~(size_t)255; return p;
  };
  u16* xA = (u16*)alloc((size_t)NTOK * HID * 2);
  u16* xB = (u16*)alloc((size_t)NTOK * HID * 2);
  u16* xC = (u16*)alloc((size_t)NTOK * HID * 2);
  u16* hB = (u16*)alloc((size_t)NTOK * HID * 2);
  float* sSrc = (float*)alloc((size_t)NTOK * 4 * 4);
  float* sDst = (float*)alloc((size_t)NTOK * 4 * 4);
  float* teE = (float*)alloc(7 * 256 * 4);   float* teS = (float*)alloc(7 * 4 * 4);
  float* meE = (float*)alloc(65 * 128 * 4);  float* meS = (float*)alloc(65 * 4 * 4);
  u16* msgBf  = (u16*)alloc(MSGS * HID * 2);
  u16* kBf    = (u16*)alloc(MSGS * HID * 2);
  u16* vBf    = (u16*)alloc(MSGS * HID * 2);
  u16* pooled = (u16*)alloc(MSGS * HID * 2);
  u16* msgOut = (u16*)alloc(MSGS * DMSGD * 2);
  u16* feat   = (u16*)alloc(MSGS * DMSGD * 2);
  u16* mh     = (u16*)alloc(MSGS * DMSGD * 2);
  u16* y0     = (u16*)alloc(MSGS * DMSGD * 2);
  u16* btT0  = (u16*)alloc(256 * 128 * 2);
  u16* btT1  = (u16*)alloc(256 * 256 * 2);
  u16* btWq  = (u16*)alloc(256 * 256 * 2);
  u16* btWk  = (u16*)alloc(256 * 256 * 2);
  u16* btWv  = (u16*)alloc(256 * 256 * 2);
  u16* btWo  = (u16*)alloc(256 * 256 * 2);
  u16* btWk2 = (u16*)alloc(256 * 256 * 2);
  u16* btWv2 = (u16*)alloc(256 * 256 * 2);
  u16* btWo2 = (u16*)alloc(128 * 256 * 2);
  u16* btM0  = (u16*)alloc(128 * 128 * 2);
  u16* btM1  = (u16*)alloc(128 * 128 * 2);

  auto convT = [&](const float* W, u16* BT, int K, int C) {
    int n = K * C;
    convert_wT<<<(n + 255) / 256, 256, 0, stream>>>(W, BT, K, C);
  };
  auto gemm = [&](const u16* A, const u16* BT, const float* bias, const u16* resid,
                  u16* out, int R, int K, int C) {
    int tiles = (R / 16) * (C / 64);
    if (K == 128)
      gemm_bf16_wmma<128><<<(tiles + 3) / 4, 128, 0, stream>>>(A, BT, bias, resid, out, R, C);
    else
      gemm_bf16_wmma<256><<<(tiles + 3) / 4, 128, 0, stream>>>(A, BT, bias, resid, out, R, C);
  };

  // ---- one-time weight transposes to bf16 ---------------------------------
  convT(F(IN_T0_W), btT0, 128, 256);  convT(F(IN_T1_W), btT1, 256, 256);
  convT(F(IN_CR_WQ), btWq, 256, 256); convT(F(IN_CR_WK), btWk, 256, 256);
  convT(F(IN_CR_WV), btWv, 256, 256); convT(F(IN_CR_WO), btWo, 256, 256);
  convT(F(IN_WK2), btWk2, 256, 256);  convT(F(IN_WV2), btWv2, 256, 256);
  convT(F(IN_WO2), btWo2, 256, 128);
  convT(F(IN_M0_W), btM0, 128, 128);  convT(F(IN_M1_W), btM1, 128, 128);

  // ---- embedding ----------------------------------------------------------
  embed_kernel<<<NTOK / 8, 256, 0, stream>>>(F(IN_TOK_EMB), F(IN_POS_EMB), F(IN_SEG_EMB),
      F(IN_LN_EMB_S), F(IN_LN_EMB_B), I(IN_TOKEN_IDS), I(IN_SEGMENT_IDS), xA);

  // ---- token GAT layer 0 (no residual) ------------------------------------
  gemm(xA, btT0, F(IN_T0_B), nullptr, hB, NTOK, 128, 256);
  attn_score_kernel<<<NTOK / 8, 256, 0, stream>>>(hB, F(IN_T0_ASRC), F(IN_T0_ADST), sSrc, sDst, NTOK, 256);
  edge_embed_kernel<<<7, 256, 0, stream>>>(F(IN_TOK_EDGE_EMB), F(IN_T0_WE), F(IN_T0_AEDGE), teE, teS, 256);
  gat_stencil_kernel<256, 3><<<NTOK / 8, 256, 0, stream>>>(hB, sSrc, sDst, teE, teS, nullptr,
      F(IN_T0_LNS), F(IN_T0_LNB), xB, nullptr, NTOK, SEQ, 3);

  // ---- token GAT layer 1 (residual) ---------------------------------------
  gemm(xB, btT1, F(IN_T1_B), nullptr, hB, NTOK, 256, 256);
  attn_score_kernel<<<NTOK / 8, 256, 0, stream>>>(hB, F(IN_T1_ASRC), F(IN_T1_ADST), sSrc, sDst, NTOK, 256);
  edge_embed_kernel<<<7, 256, 0, stream>>>(F(IN_TOK_EDGE_EMB), F(IN_T1_WE), F(IN_T1_AEDGE), teE, teS, 256);
  gat_stencil_kernel<256, 3><<<NTOK / 8, 256, 0, stream>>>(hB, sSrc, sDst, teE, teS, xB,
      F(IN_T1_LNS), F(IN_T1_LNB), xA, nullptr, NTOK, SEQ, 3);

  // ---- cross attention ----------------------------------------------------
  seq_mean_kernel<<<MSGS, 256, 0, stream>>>(xA, msgBf);
  gemm(xA, btWq, nullptr, nullptr, hB, NTOK, 256, 256);          // q
  gemm(msgBf, btWk, nullptr, nullptr, kBf, MSGS, 256, 256);      // k
  gemm(msgBf, btWv, nullptr, nullptr, vBf, MSGS, 256, 256);      // v
  cross_topk_kernel<<<1024, 256, 0, stream>>>(hB, kBf, vBf, xC, NTOK);
  gemm(xC, btWo, nullptr, xA, xB, NTOK, 256, 256);               // x + attn@Wo

  // ---- token -> message pooling -------------------------------------------
  gemm(xB, btWk2, nullptr, nullptr, hB, NTOK, 256, 256);         // k2
  gemm(xB, btWv2, nullptr, nullptr, xC, NTOK, 256, 256);         // v2
  t2m_kernel<<<MSGS, 128, 0, stream>>>(hB, xC, F(IN_Q2), F(IN_POSW), F(IN_POSB), pooled);
  gemm(pooled, btWo2, F(IN_BO2), nullptr, msgOut, MSGS, 256, 128);

  // ---- encoder ------------------------------------------------------------
  enc_kernel<<<MSGS / 8, 256, 0, stream>>>(msgOut, F(IN_MSG_ATTR), F(IN_ENC_W), F(IN_ENC_B),
      F(IN_ENC_LNS), F(IN_ENC_LNB), feat);

  // ---- message GAT layer 0 (chain graph, relBase=32) ----------------------
  gemm(feat, btM0, F(IN_M0_B), nullptr, mh, MSGS, 128, 128);
  attn_score_kernel<<<MSGS / 8, 256, 0, stream>>>(mh, F(IN_M0_ASRC), F(IN_M0_ADST), sSrc, sDst, MSGS, 128);
  edge_embed_kernel<<<65, 128, 0, stream>>>(F(IN_MSG_RELPOS), F(IN_M0_WE), F(IN_M0_AEDGE), meE, meS, 128);
  gat_stencil_kernel<128, 1><<<MSGS / 8, 256, 0, stream>>>(mh, sSrc, sDst, meE, meS, nullptr,
      F(IN_M0_LNS), F(IN_M0_LNB), y0, nullptr, MSGS, MSGS, 32);

  // ---- message GAT layer 1 (residual) -> f32 output -----------------------
  gemm(y0, btM1, F(IN_M1_B), nullptr, mh, MSGS, 128, 128);
  attn_score_kernel<<<MSGS / 8, 256, 0, stream>>>(mh, F(IN_M1_ASRC), F(IN_M1_ADST), sSrc, sDst, MSGS, 128);
  edge_embed_kernel<<<65, 128, 0, stream>>>(F(IN_MSG_RELPOS), F(IN_M1_WE), F(IN_M1_AEDGE), meE, meS, 128);
  gat_stencil_kernel<128, 1><<<MSGS / 8, 256, 0, stream>>>(mh, sSrc, sDst, meE, meS, y0,
      F(IN_M1_LNS), F(IN_M1_LNB), nullptr, (float*)d_out, MSGS, MSGS, 32);
}
